// SimilarityLoss_86492051407493
// MI455X (gfx1250) — matrix operand order repaired and checked by reading, not compiled, not run on hardware
//
#include <hip/hip_runtime.h>
#include <hip/hip_bf16.h>

typedef __attribute__((ext_vector_type(8)))  _Float16 v8h;
typedef __attribute__((ext_vector_type(16))) _Float16 v16h;
typedef __attribute__((ext_vector_type(8)))  float    v8f;
typedef __attribute__((ext_vector_type(4)))  unsigned int u32x4;
typedef __attribute__((ext_vector_type(8)))  int      i32x8;
typedef __attribute__((ext_vector_type(4)))  int      i32x4;

#define B_ROWS   4096
#define DIM      256
#define NROWS    8192            // 2B
#define TEMP_INV 2.0f            // 1/T, T = 0.5
#define EXP2_SCALE 2.885390082f  // 2 * log2(e): exp(2x) = exp2(x * this)
#define EPSN     1e-8f

#define WAVES_PER_BLOCK 8
#define BLOCK_THREADS   256      // 8 wave32s
#define ROW_TILE        128      // rows per workgroup (8 waves x 16)
#define COL_TILE        128      // columns staged in LDS per iteration
#define COL_SPLITS      8        // grid.y; 8192/8 = 1024 cols per split
#define N_TILES         ((NROWS / COL_SPLITS) / COL_TILE)   // 8
#define LDS_COL_STRIDE  272      // 256 + 16 halves pad (544B = TDM pad 32B per 512B)
#define BUF_HALVES      (COL_TILE * LDS_COL_STRIDE)         // one buffer, in halves

// TDM (tensor_load_to_lds) availability: device pass only; arity differs per toolchain.
#if defined(__HIP_DEVICE_COMPILE__) && __has_builtin(__builtin_amdgcn_tensor_load_to_lds)
#define USE_TDM 1
#else
#define USE_TDM 0
#endif

// ---------------- kernel 1: normalize rows -> f16, and f16 self-dot ----------------
__global__ void prep_kernel(const float* __restrict__ z1, const float* __restrict__ z2,
                            _Float16* __restrict__ znf16, float* __restrict__ selfdot) {
  const int wave = threadIdx.x >> 5;
  const int lane = threadIdx.x & 31;
  const int row  = blockIdx.x * WAVES_PER_BLOCK + wave;
  const float* src = (row < B_ROWS) ? (z1 + (size_t)row * DIM)
                                    : (z2 + (size_t)(row - B_ROWS) * DIM);
  const float4* s4 = (const float4*)(src + lane * 8);
  float4 a0 = s4[0], a1 = s4[1];
  float x[8] = {a0.x, a0.y, a0.z, a0.w, a1.x, a1.y, a1.z, a1.w};
  float ss = 0.f;
#pragma unroll
  for (int j = 0; j < 8; ++j) ss += x[j] * x[j];
#pragma unroll
  for (int m = 16; m >= 1; m >>= 1) ss += __shfl_xor(ss, m, 32);
  const float inv = 1.0f / fmaxf(sqrtf(ss), EPSN);
  v8h hv;
  float sd = 0.f;
#pragma unroll
  for (int j = 0; j < 8; ++j) {
    _Float16 h = (_Float16)(x[j] * inv);
    hv[j] = h;
    float f = (float)h;
    sd += f * f;
  }
  *(v8h*)(znf16 + (size_t)row * DIM + lane * 8) = hv;
#pragma unroll
  for (int m = 16; m >= 1; m >>= 1) sd += __shfl_xor(sd, m, 32);
  if (lane == 0) selfdot[row] = sd;
}

// ---------------- kernel 2: exact fp32 positives pos[i] = cos(z1_i, z2_i) ----------------
__global__ void pos_kernel(const float* __restrict__ z1, const float* __restrict__ z2,
                           float* __restrict__ posbuf) {
  const int wave = threadIdx.x >> 5;
  const int lane = threadIdx.x & 31;
  const int i = blockIdx.x * WAVES_PER_BLOCK + wave;
  const float4* p1 = (const float4*)(z1 + (size_t)i * DIM + lane * 8);
  const float4* p2 = (const float4*)(z2 + (size_t)i * DIM + lane * 8);
  float4 a0 = p1[0], a1 = p1[1], b0 = p2[0], b1 = p2[1];
  float xa[8] = {a0.x, a0.y, a0.z, a0.w, a1.x, a1.y, a1.z, a1.w};
  float xb[8] = {b0.x, b0.y, b0.z, b0.w, b1.x, b1.y, b1.z, b1.w};
  float s11 = 0.f, s22 = 0.f, s12 = 0.f;
#pragma unroll
  for (int j = 0; j < 8; ++j) {
    s11 += xa[j] * xa[j];
    s22 += xb[j] * xb[j];
    s12 += xa[j] * xb[j];
  }
#pragma unroll
  for (int m = 16; m >= 1; m >>= 1) {
    s11 += __shfl_xor(s11, m, 32);
    s22 += __shfl_xor(s22, m, 32);
    s12 += __shfl_xor(s12, m, 32);
  }
  if (lane == 0)
    posbuf[i] = s12 / (fmaxf(sqrtf(s11), EPSN) * fmaxf(sqrtf(s22), EPSN));
}

#if USE_TDM
// Issue one TDM load: 128 contiguous rows x 512B from global into LDS, padded to
// 544B/row in LDS (pad_interval = 512B, pad_amount = 8 DWORDs = 32B).  D# per ISA §8.
__device__ __forceinline__ void tdm_load_tile(unsigned lds_byte_off,
                                              const _Float16* gsrc) {
  unsigned long long ga = (unsigned long long)(uintptr_t)gsrc;
  u32x4 g0 = {};
  g0[0] = 1u;                                      // count=1, user descriptor
  g0[1] = lds_byte_off;                            // lds_addr [63:32]
  g0[2] = (unsigned)(ga & 0xFFFFFFFFu);            // global_addr [95:64]
  g0[3] = (unsigned)((ga >> 32) & 0x01FFFFFFu)     // global_addr[56:32] -> [120:96]
        | (2u << 30);                              // type = 2 ("image")
  i32x8 g1 = {};
  g1[0] = (3 << 16)                                // data_size = 8 bytes
        | (1 << 20)                                // pad_enable
        | (6 << 22)                                // pad_interval: 2^(6+1)=128 DW = 512B
        | (7 << 25);                               // pad_amount: 8 DWORDs = 32B
  g1[1] = (int)(64u << 16);                        // tensor_dim0 = 64 (8B units)
  g1[2] = (int)(0x2000u << 16);                    // tensor_dim1 = 8192 rows
  g1[3] = (int)(64u << 16);                        // tile_dim0 = 64 (8B units = 512B)
  g1[4] = COL_TILE;                                // tile_dim1 = 128 rows
  g1[5] = 64;                                      // tensor_dim0_stride = 64 (8B units)
  i32x4 gz = {};
#if __clang_major__ >= 23
  i32x8 gz8 = {};
  __builtin_amdgcn_tensor_load_to_lds(g0, g1, gz, gz, gz8, 0);
#else
  __builtin_amdgcn_tensor_load_to_lds(g0, g1, gz, gz, 0);
#endif
}
#endif

// ---- explicit software pipeline pieces for one 16x16 subtile -------------------
__device__ __forceinline__ void load_b(const _Float16* __restrict__ bcol, v16h* Bf) {
  // B 32x16 f16 layout: lane holds column (lane&15), contiguous K in
  // [khalf*16, khalf*16+16) per K32 chunk; 16 ds_load_b128 issued as a burst.
#pragma unroll
  for (int q = 0; q < 8; ++q) Bf[q] = *(const v16h*)(bcol + q * 32);
}

__device__ __forceinline__ v8f wmma_chain(const v16h* A, const v16h* Bf) {
  v8f c = {};
#pragma unroll
  for (int q = 0; q < 8; ++q)
    c = __builtin_amdgcn_wmma_f32_16x16x32_f16(
        /*neg_a=*/false, A[q], /*neg_b=*/false, Bf[q],
        /*c_mod=*/(short)0, c, /*reuse_a=*/false, /*reuse_b=*/false);
  return c;
}

__device__ __forceinline__ void exp_epilogue(const v8f& c, float* rs) {
  // C layout: c[v] = sim[rowbase + khalf*8 + v, coltile + n]; exp(sim/T) folded in.
#pragma unroll
  for (int v = 0; v < 8; ++v)
    rs[v] += __builtin_amdgcn_exp2f(c[v] * EXP2_SCALE);
}

// 128-column LDS buffer: pipeline = {load B[sub]} || {exp epilogue of sub-1}, then WMMA sub
__device__ __forceinline__ void tile_accum(const _Float16* __restrict__ buf,
                                           const v16h* A, float* rs,
                                           int n, int khalf) {
  const _Float16* base = buf + n * LDS_COL_STRIDE + khalf * 16;
  v16h Bf[8];
  load_b(base, Bf);
  v8f c = wmma_chain(A, Bf);
#pragma unroll
  for (int sub = 1; sub < COL_TILE / 16; ++sub) {
    load_b(base + sub * 16 * LDS_COL_STRIDE, Bf);   // next subtile's LDS loads in flight
    exp_epilogue(c, rs);                            // VALU/TRANS covers LDS latency
    c = wmma_chain(A, Bf);
  }
  exp_epilogue(c, rs);
}

// ---------------- kernel 3: fused zn@zn^T -> exp row-sums (TDM + WMMA) ----------------
__global__ void __launch_bounds__(BLOCK_THREADS)
simexp_kernel(const _Float16* __restrict__ znf16, float* __restrict__ partial) {
  extern __shared__ __align__(32) _Float16 lds[];   // 2 x BUF_HALVES (double buffer)
  const int wave     = threadIdx.x >> 5;
  const int lane     = threadIdx.x & 31;
  const int n        = lane & 15;
  const int khalf    = lane >> 4;
  const int rowbase  = blockIdx.x * ROW_TILE + wave * 16;
  const int colsplit = blockIdx.y;

  // Resident A fragments: this wave's 16 rows, full K=256.
  // A 16x32 f16 layout: lane holds row (lane&15); contiguous K in
  // [khalf*16, khalf*16+16) per K32 chunk (matches B).
  v16h A[8];
  {
    const _Float16* arow = znf16 + (size_t)(rowbase + n) * DIM;
#pragma unroll
    for (int q = 0; q < 8; ++q) {
      v8h lo = *(const v8h*)(arow + q * 32 + khalf * 16);
      v8h hi = *(const v8h*)(arow + q * 32 + khalf * 16 + 8);
      A[q] = __builtin_shufflevector(lo, hi, 0, 1, 2, 3, 4, 5, 6, 7,
                                     8, 9, 10, 11, 12, 13, 14, 15);
    }
  }

  float rs[8] = {0.f, 0.f, 0.f, 0.f, 0.f, 0.f, 0.f, 0.f};
  const int col0 = colsplit * (NROWS / COL_SPLITS);
  const _Float16* gbase = znf16 + (size_t)col0 * DIM;   // 128-row tiles, 512B rows

#if USE_TDM
  // Double-buffered TDM pipeline: wave 0 drives the Tensor Data Mover.
  if (wave == 0) {
    tdm_load_tile((unsigned)(uintptr_t)lds, gbase);
    __builtin_amdgcn_s_wait_tensorcnt(0);
  }
  __syncthreads();
  for (int ct = 0; ct < N_TILES; ++ct) {
    const int cur = ct & 1;
    if (wave == 0 && (ct + 1) < N_TILES)
      tdm_load_tile((unsigned)(uintptr_t)(lds + (size_t)((ct + 1) & 1) * BUF_HALVES),
                    gbase + (size_t)(ct + 1) * COL_TILE * DIM);
    tile_accum(lds + (size_t)cur * BUF_HALVES, A, rs, n, khalf);  // overlaps with TDM
    if (wave == 0) __builtin_amdgcn_s_wait_tensorcnt(0);
    __syncthreads();   // next tile visible to all; prev buffer free for reuse
  }
#else
  // Fallback: synchronous cooperative staging into buffer 0.
  for (int ct = 0; ct < N_TILES; ++ct) {
    __syncthreads();
    for (int u = threadIdx.x; u < COL_TILE * 16; u += BLOCK_THREADS) {
      const int c = u >> 4, seg = u & 15;
      v16h v = *(const v16h*)(gbase + (size_t)(ct * COL_TILE + c) * DIM + seg * 16);
      *(v16h*)(lds + c * LDS_COL_STRIDE + seg * 16) = v;
    }
    __syncthreads();
    tile_accum(lds, A, rs, n, khalf);
  }
#endif

  // Reduce across the 16 lanes sharing khalf (same rows, different columns).
#pragma unroll
  for (int v = 0; v < 8; ++v) {
    float s = rs[v];
#pragma unroll
    for (int m = 8; m >= 1; m >>= 1) s += __shfl_xor(s, m, 16);
    rs[v] = s;
  }
  if (n == 0) {
    const int r0 = rowbase + khalf * 8;
#pragma unroll
    for (int v = 0; v < 8; ++v)
      partial[(size_t)(r0 + v) * COL_SPLITS + colsplit] = rs[v];
  }
}

// ---------------- kernel 4: deterministic final reduction -> scalar loss ----------------
__global__ void finalize_kernel(const float* __restrict__ partial,
                                const float* __restrict__ selfdot,
                                const float* __restrict__ posbuf,
                                float* __restrict__ out) {
  __shared__ float wsum[WAVES_PER_BLOCK];
  float local = 0.f;
  for (int r = threadIdx.x; r < NROWS; r += BLOCK_THREADS) {
    float s = 0.f;
#pragma unroll
    for (int k = 0; k < COL_SPLITS; ++k) s += partial[(size_t)r * COL_SPLITS + k];
    const float denom = s - __expf(selfdot[r] * TEMP_INV);  // mask self-similarity
    const float pos = posbuf[r & (B_ROWS - 1)];             // r % B
    local += __logf(pos) - __logf(denom);
  }
#pragma unroll
  for (int m = 16; m >= 1; m >>= 1) local += __shfl_xor(local, m, 32);
  const int wave = threadIdx.x >> 5, lane = threadIdx.x & 31;
  if (lane == 0) wsum[wave] = local;
  __syncthreads();
  if (threadIdx.x == 0) {
    float t = 0.f;
#pragma unroll
    for (int w = 0; w < WAVES_PER_BLOCK; ++w) t += wsum[w];
    out[0] = -t / (float)NROWS;
  }
}

extern "C" void kernel_launch(void* const* d_in, const int* in_sizes, int n_in,
                              void* d_out, int out_size, void* d_ws, size_t ws_size,
                              hipStream_t stream) {
  (void)in_sizes; (void)n_in; (void)out_size; (void)ws_size;
  const float* z1 = (const float*)d_in[0];
  const float* z2 = (const float*)d_in[1];
  float* out = (float*)d_out;

  // Workspace carve-out (hipMalloc'd base => 256B aligned).
  char* ws = (char*)d_ws;
  _Float16* znf16 = (_Float16*)ws;                                   // 4 MiB
  size_t off = (size_t)NROWS * DIM * sizeof(_Float16);
  float* partial = (float*)(ws + off);                               // 256 KiB
  off += (size_t)NROWS * COL_SPLITS * sizeof(float);
  float* selfdot = (float*)(ws + off);                               // 32 KiB
  off += (size_t)NROWS * sizeof(float);
  float* posbuf = (float*)(ws + off);                                // 16 KiB

  prep_kernel<<<NROWS / WAVES_PER_BLOCK, BLOCK_THREADS, 0, stream>>>(z1, z2, znf16, selfdot);
  pos_kernel<<<B_ROWS / WAVES_PER_BLOCK, BLOCK_THREADS, 0, stream>>>(z1, z2, posbuf);

  dim3 grid(NROWS / ROW_TILE, COL_SPLITS);
  // Always size for the double buffer so host/device paths can't disagree.
  size_t ldsbytes = 2 * (size_t)BUF_HALVES * sizeof(_Float16);       // 139,264 B
  simexp_kernel<<<grid, BLOCK_THREADS, ldsbytes, stream>>>(znf16, partial);

  finalize_kernel<<<1, BLOCK_THREADS, 0, stream>>>(partial, selfdot, posbuf, out);
}
